// Phoneme_GAT_36610301231358
// MI455X (gfx1250) — compile-verified, matrix operand order checked by reading, not run on hardware
//
#include <hip/hip_runtime.h>
#include <hip/hip_bf16.h>

#define NNODES 60000
#define NEDGES 240000
#define FTOT   768      // F_IN and H*F_OUT (both 768)
#define HEADS  6
#define FOUT   128

// wave tile for the WMMA GEMM: 48 rows x 64 cols = 3x4 tiles of 16x16.
// 12 accumulators (96 VGPRs) + 7 fragments (56 VGPRs) needs the
// full 256-VGPR window -> __launch_bounds__(256, 1) to lift the compiler's
// default 128-VGPR occupancy cap (which caused accumulator spills to scratch).
#define TM 3
#define TN 4

typedef __attribute__((ext_vector_type(16))) __bf16 v16bf;
typedef __attribute__((ext_vector_type(8)))  float  v8f;

union FragU { uint4 q[2]; v16bf v; };

// ---------------- helpers ----------------
__device__ __forceinline__ int floatFlip(float f) {
    int i = __float_as_int(f);
    return (i >= 0) ? i : (i ^ 0x7FFFFFFF);   // order-preserving map for signed atomicMax
}
__device__ __forceinline__ float floatUnflip(int i) {
    int j = (i >= 0) ? i : (i ^ 0x7FFFFFFF);
    return __int_as_float(j);
}

// ---------------- conversion kernels ----------------
__global__ void k_f32_to_bf16(const float* __restrict__ in, __hip_bfloat16* __restrict__ out, int n) {
    int i = blockIdx.x * blockDim.x + threadIdx.x;
    if (i < n) out[i] = __float2bfloat16(in[i]);
}

// W is (K x N) row-major f32; produce Wt (N x K) row-major bf16
__global__ void k_transpose_bf16(const float* __restrict__ W, __hip_bfloat16* __restrict__ Wt,
                                 int K, int N) {
    int i = blockIdx.x * blockDim.x + threadIdx.x;
    if (i >= K * N) return;
    int n = i / K, k = i % K;
    Wt[i] = __float2bfloat16(W[k * N + n]);
}

// ---------------- WMMA bf16 GEMM: C(MxN,f32) = A(MxK,bf16) * B(KxN) with B given transposed ----------
// one wave -> 48x64 output block; 3x4 tiles, 12 v_wmma per K-step, 7 fragment loads per K-step.
// Requires M % 48 == 0, N % 64 == 0, K % 32 == 0 (all hold: 60000/48=1250, 768/64=12, 768/32=24).
__global__ void __launch_bounds__(256, 1)
k_gemm_bf16_wmma(const __hip_bfloat16* __restrict__ A,
                 const __hip_bfloat16* __restrict__ Bt,   // N x K row-major
                 float* __restrict__ C,
                 int M, int N, int K) {
    const int lane  = threadIdx.x & 31;
    const int wib   = threadIdx.x >> 5;
    const int wpb   = blockDim.x >> 5;
    const int wave  = blockIdx.x * wpb + wib;
    const int tilesN = N / (TN * 16);
    const int totalWaves = (M / (TM * 16)) * tilesN;
    if (wave >= totalWaves) return;               // wave-uniform

    const int tm = (wave / tilesN) * (TM * 16);
    const int tn = (wave % tilesN) * (TN * 16);

    const int r  = lane & 15;                     // row-in-tile for A / col-in-tile for B
    const int kb = (lane >> 4) << 3;              // K base 0 or 8 per ISA 16-bit A/B layout

    v8f acc[TM][TN] = {};

    const __hip_bfloat16* pa0 = A  + (size_t)(tm + r) * K + kb;
    const __hip_bfloat16* pb0 = Bt + (size_t)(tn + r) * K + kb;

    for (int k0 = 0; k0 < K; k0 += 32) {
        FragU a[TM], b[TN];
#pragma unroll
        for (int i = 0; i < TM; ++i) {
            const __hip_bfloat16* pa = pa0 + (size_t)(i * 16) * K + k0;
            a[i].q[0] = *(const uint4*)(pa);
            a[i].q[1] = *(const uint4*)(pa + 16);
        }
#pragma unroll
        for (int j = 0; j < TN; ++j) {
            const __hip_bfloat16* pb = pb0 + (size_t)(j * 16) * K + k0;
            b[j].q[0] = *(const uint4*)(pb);
            b[j].q[1] = *(const uint4*)(pb + 16);
        }
#pragma unroll
        for (int i = 0; i < TM; ++i)
#pragma unroll
            for (int j = 0; j < TN; ++j)
                acc[i][j] = __builtin_amdgcn_wmma_f32_16x16x32_bf16(
                    false, a[i].v, false, b[j].v, (short)0, acc[i][j], false, false);
    }

    // C/D layout: vgpr v -> row = v + 8*(lane>=16); col = lane&15
    const int rowOff = (lane >> 4) << 3;
    const int col    = lane & 15;
#pragma unroll
    for (int i = 0; i < TM; ++i)
#pragma unroll
        for (int j = 0; j < TN; ++j) {
            float* cp = C + (size_t)(tm + i * 16 + rowOff) * N + tn + j * 16 + col;
#pragma unroll
            for (int v = 0; v < 8; ++v)
                cp[(size_t)v * N] = acc[i][j][v];
        }
}

// ---------------- attention scores: s[n][h] = <proj[n,h,:], a[h,:]> ----------------
__global__ void k_scores(const float* __restrict__ proj,
                         const float* __restrict__ a_src, const float* __restrict__ a_tgt,
                         float* __restrict__ s_src, float* __restrict__ s_tgt, int Nn) {
    const int wave = (blockIdx.x * blockDim.x + threadIdx.x) >> 5;
    const int lane = threadIdx.x & 31;
    if (wave >= Nn * HEADS) return;
    const int n = wave / HEADS, h = wave % HEADS;
    const float* p  = proj  + (size_t)n * FTOT + h * FOUT;
    const float* as = a_src + h * FOUT;
    const float* at = a_tgt + h * FOUT;
    float ss = 0.f, st = 0.f;
    for (int f = lane; f < FOUT; f += 32) { float v = p[f]; ss += v * as[f]; st += v * at[f]; }
#pragma unroll
    for (int o = 16; o; o >>= 1) { ss += __shfl_xor(ss, o, 32); st += __shfl_xor(st, o, 32); }
    if (lane == 0) { s_src[wave] = ss; s_tgt[wave] = st; }
}

__global__ void k_init_nodes(int* __restrict__ nmax, float* __restrict__ nsum, int n) {
    int i = blockIdx.x * blockDim.x + threadIdx.x;
    if (i < n) { nmax[i] = (int)0x80000000; nsum[i] = 0.f; }
}

// pass 1: leaky-relu logits + segment max
__global__ void k_edge_logits(const int* __restrict__ ei, const float* __restrict__ s_src,
                              const float* __restrict__ s_tgt, float* __restrict__ eV,
                              int* __restrict__ nmax) {
    int idx = blockIdx.x * blockDim.x + threadIdx.x;
    if (idx >= NEDGES * HEADS) return;
    int e = idx / HEADS, h = idx % HEADS;
    int s = ei[e], t = ei[NEDGES + e];
    float v = s_src[s * HEADS + h] + s_tgt[t * HEADS + h];
    v = (v > 0.f) ? v : 0.2f * v;
    eV[idx] = v;
    atomicMax(&nmax[t * HEADS + h], floatFlip(v));
}

// pass 2: exp + segment sum
__global__ void k_edge_exp(const int* __restrict__ ei, float* __restrict__ eV,
                           const int* __restrict__ nmax, float* __restrict__ nsum) {
    int idx = blockIdx.x * blockDim.x + threadIdx.x;
    if (idx >= NEDGES * HEADS) return;
    int e = idx / HEADS, h = idx % HEADS;
    int t = ei[NEDGES + e];
    float m  = floatUnflip(nmax[t * HEADS + h]);
    float ex = __expf(eV[idx] - m);
    eV[idx] = ex;
    atomicAdd(&nsum[t * HEADS + h], ex);
}

// pass 3: messages, wave per edge; acc pre-initialized with skip projection
__global__ void k_edge_messages(const int* __restrict__ ei, const float* __restrict__ eV,
                                const float* __restrict__ nsum, const float* __restrict__ proj,
                                float* __restrict__ acc) {
    const int e = blockIdx.x * (blockDim.x >> 5) + (threadIdx.x >> 5);
    if (e >= NEDGES) return;
    const int lane = threadIdx.x & 31;
    const int s = ei[e], t = ei[NEDGES + e];
    float alpha[HEADS];
#pragma unroll
    for (int h = 0; h < HEADS; ++h)
        alpha[h] = eV[e * HEADS + h] / (nsum[t * HEADS + h] + 1e-16f);
    const float* ps = proj + (size_t)s * FTOT;
    float*       pa = acc  + (size_t)t * FTOT;
    for (int c = lane; c < FTOT; c += 32)
        atomicAdd(&pa[c], ps[c] * alpha[c >> 7]);   // c>>7 == head (FOUT==128)
}

// layers 0/1: out = elu(acc + bias) ; concat layout already contiguous
__global__ void k_finalize_elu(const float* __restrict__ acc, const float* __restrict__ bias,
                               float* __restrict__ out, int total) {
    int i = blockIdx.x * blockDim.x + threadIdx.x;
    if (i >= total) return;
    float v = acc[i] + bias[i % FTOT];
    out[i] = (v > 0.f) ? v : (__expf(v) - 1.f);
}

// layer 2: out = mean_h(acc) + bias, no activation
__global__ void k_finalize_mean(const float* __restrict__ acc, const float* __restrict__ bias,
                                float* __restrict__ out, int Nn) {
    int i = blockIdx.x * blockDim.x + threadIdx.x;
    if (i >= Nn * FOUT) return;
    int n = i / FOUT, f = i % FOUT;
    float s = 0.f;
#pragma unroll
    for (int h = 0; h < HEADS; ++h) s += acc[(size_t)n * FTOT + h * FOUT + f];
    out[i] = s * (1.f / 6.f) + bias[f];
}

// ---------------- host driver ----------------
static inline int ceilDiv(int a, int b) { return (a + b - 1) / b; }

extern "C" void kernel_launch(void* const* d_in, const int* in_sizes, int n_in,
                              void* d_out, int out_size, void* d_ws, size_t ws_size,
                              hipStream_t stream) {
    const float* x  = (const float*)d_in[0];
    const int*   ei = (const int*)d_in[1];
    const float* Wl[3] = {(const float*)d_in[2],  (const float*)d_in[7],  (const float*)d_in[12]};
    const float* aS[3] = {(const float*)d_in[3],  (const float*)d_in[8],  (const float*)d_in[13]};
    const float* aT[3] = {(const float*)d_in[4],  (const float*)d_in[9],  (const float*)d_in[14]};
    const float* Sk[3] = {(const float*)d_in[5],  (const float*)d_in[10], (const float*)d_in[15]};
    const float* bl[3] = {(const float*)d_in[6],  (const float*)d_in[11], (const float*)d_in[16]};

    char* ws = (char*)d_ws;
    size_t off = 0;
    auto carve = [&](size_t bytes) { void* p = ws + off; off = (off + bytes + 255) & ~(size_t)255; return p; };

    float*          h_cur  = (float*)          carve((size_t)NNODES * FTOT * 4);
    __hip_bfloat16* xb     = (__hip_bfloat16*) carve((size_t)NNODES * FTOT * 2);
    float*          proj   = (float*)          carve((size_t)NNODES * FTOT * 4);
    float*          acc    = (float*)          carve((size_t)NNODES * FTOT * 4);
    __hip_bfloat16* Wt     = (__hip_bfloat16*) carve((size_t)FTOT * FTOT * 2);
    __hip_bfloat16* St     = (__hip_bfloat16*) carve((size_t)FTOT * FTOT * 2);
    float*          s_src  = (float*)          carve((size_t)NNODES * HEADS * 4);
    float*          s_tgt  = (float*)          carve((size_t)NNODES * HEADS * 4);
    int*            nmax   = (int*)            carve((size_t)NNODES * HEADS * 4);
    float*          nsum   = (float*)          carve((size_t)NNODES * HEADS * 4);
    float*          eV     = (float*)          carve((size_t)NEDGES * HEADS * 4);
    (void)ws_size;

    const int T = 256;
    const int nElems     = NNODES * FTOT;                               // 46,080,000
    const int gemmWaves  = (NNODES / (TM * 16)) * (FTOT / (TN * 16));   // 1250*12 = 15000
    const int gemmBlocks = ceilDiv(gemmWaves * 32, T);                  // 1875
    const int scoreBlocks = ceilDiv(NNODES * HEADS * 32, T);
    const int ehBlocks    = ceilDiv(NEDGES * HEADS, T);
    const int msgBlocks   = ceilDiv(NEDGES, T / 32);

    for (int l = 0; l < 3; ++l) {
        const float* in_f32 = (l == 0) ? x : h_cur;

        k_f32_to_bf16<<<ceilDiv(nElems, T), T, 0, stream>>>(in_f32, xb, nElems);
        k_transpose_bf16<<<ceilDiv(FTOT * FTOT, T), T, 0, stream>>>(Wl[l], Wt, FTOT, FTOT);
        k_transpose_bf16<<<ceilDiv(FTOT * FTOT, T), T, 0, stream>>>(Sk[l], St, FTOT, FTOT);

        k_gemm_bf16_wmma<<<gemmBlocks, T, 0, stream>>>(xb, Wt, proj, NNODES, FTOT, FTOT);
        k_gemm_bf16_wmma<<<gemmBlocks, T, 0, stream>>>(xb, St, acc,  NNODES, FTOT, FTOT);  // acc = skip proj

        k_scores<<<scoreBlocks, T, 0, stream>>>(proj, aS[l], aT[l], s_src, s_tgt, NNODES);
        k_init_nodes<<<ceilDiv(NNODES * HEADS, T), T, 0, stream>>>(nmax, nsum, NNODES * HEADS);
        k_edge_logits<<<ehBlocks, T, 0, stream>>>(ei, s_src, s_tgt, eV, nmax);
        k_edge_exp<<<ehBlocks, T, 0, stream>>>(ei, eV, nmax, nsum);
        k_edge_messages<<<msgBlocks, T, 0, stream>>>(ei, eV, nsum, proj, acc);

        if (l < 2)
            k_finalize_elu<<<ceilDiv(nElems, T), T, 0, stream>>>(acc, bl[l], h_cur, nElems);
        else
            k_finalize_mean<<<ceilDiv(NNODES * FOUT, T), T, 0, stream>>>(acc, bl[l], (float*)d_out, NNODES);
    }
}
